// MultiHeadAttention_16423954940513
// MI455X (gfx1250) — matrix operand order repaired and checked by reading, not compiled, not run on hardware
//
#include <hip/hip_runtime.h>
#include <hip/hip_bf16.h>
#include <stdint.h>

// Problem constants: B=2, T=2048, M=2048, D=128, G=4, H=4
#define KB 2
#define KT 2048
#define KM 2048
#define KD 128
#define KG 4
#define KH 4
#define BT (KB * KT)   // 4096 token rows
#define NQKV 3072      // q: 16 heads*128 | k: 4 heads*128 | v: 4 heads*128
#define LDP 40         // padded LDS stride (elements) for 32-wide K tiles
#define LDKS 136       // padded LDS stride (elements) for 128-wide D tiles

typedef __bf16 bf16;
typedef __attribute__((ext_vector_type(16))) __bf16 bf16x16;
typedef __attribute__((ext_vector_type(8)))  float  f32x8;

union AF { bf16x16 v; bf16 e[16]; };
union CF { f32x8 v; float f[8]; };

__device__ __forceinline__ bf16 f2bf(float f) {
  union { float f; uint32_t u; } x; x.f = f;
  uint32_t r = x.u + 0x7FFFu + ((x.u >> 16) & 1u);   // round-to-nearest-even
  uint16_t h = (uint16_t)(r >> 16);
  bf16 o; __builtin_memcpy(&o, &h, 2); return o;
}
__device__ __forceinline__ float bf2f(bf16 b) {
  uint16_t h; __builtin_memcpy(&h, &b, 2);
  union { uint32_t u; float f; } x; x.u = ((uint32_t)h) << 16;
  return x.f;
}

// ---------------------------------------------------------------------------
// CDNA5 async memory->LDS copy (GLOBAL_LOAD_ASYNC_TO_LDS_B128, ASYNCcnt).
// Generic pointers to LDS carry the LDS byte offset in their low 32 bits
// (ISA 10.2 aperture rules), so we can pass that dword as VDST directly.
// ---------------------------------------------------------------------------
__device__ __forceinline__ void async_copy_b128(const void* gsrc, void* lds_dst) {
  uint32_t lds = (uint32_t)(uintptr_t)lds_dst;
  asm volatile("global_load_async_to_lds_b128 %0, %1, off"
               :: "v"(lds), "v"(gsrc) : "memory");
}
__device__ __forceinline__ void wait_async() {
  asm volatile("s_wait_asynccnt 0" ::: "memory");
}

// A fragment: 16x32 (MxK) bf16, row-major source with stride ld.
// Layout: lanes 0-15 hold rows M=lane with K 0..15; lanes 16-31
// same rows with K 16..31.
__device__ __forceinline__ bf16x16 load_frag_a(const bf16* base, int ld, int lane) {
  int m = lane & 15, khi = (lane >> 4) & 1;
  const bf16* row = base + m * ld + khi * 16;
  AF a;
#pragma unroll
  for (int e = 0; e < 8; ++e) a.e[e] = row[e];          // K = khi*16 + e
#pragma unroll
  for (int e = 0; e < 8; ++e) a.e[8 + e] = row[8 + e]; // K = khi*16 + 8 + e
  return a.v;
}

// B fragment: 32x16 (KxN) bf16, sourced from an LDS tile stored TRANSPOSED as
// Bt[n][k] with stride ld. Layout (same k order as load_frag_a): lanes 0-15 hold col N=lane, K=0..15;
// lanes 16-31 hold K=16..31. Per-lane elements are contiguous in k.
__device__ __forceinline__ bf16x16 load_frag_bt(const bf16* base, int ld, int lane) {
  int n = lane & 15, kb = (lane >> 4) << 4;
  const bf16* row = base + n * ld + kb;
  AF b;
#pragma unroll
  for (int e = 0; e < 16; ++e) b.e[e] = row[e];
  return b.v;
}

__device__ __forceinline__ f32x8 wmma_bf16(bf16x16 a, bf16x16 b, f32x8 c) {
  return __builtin_amdgcn_wmma_f32_16x16x32_bf16(false, a, false, b, (short)0, c,
                                                 false, false);
}

// ---------------------------------------------------------------------------
// Conversion / packing kernels
// ---------------------------------------------------------------------------
__global__ void cvt_f32_bf16(const float* __restrict__ in, bf16* __restrict__ out, int n) {
  int i = blockIdx.x * blockDim.x + threadIdx.x;
  if (i < n) out[i] = f2bf(in[i]);
}

__global__ void pack_wqkv(const float* __restrict__ wq, const float* __restrict__ wk,
                          const float* __restrict__ wv, bf16* __restrict__ out) {
  int i = blockIdx.x * blockDim.x + threadIdx.x;   // over KM * NQKV
  if (i >= KM * NQKV) return;
  int m = i / NQKV, c = i % NQKV;
  float v;
  if (c < 2048)      v = wq[m * 2048 + c];            // [M, G*H*D]
  else if (c < 2560) v = wk[m * 512 + (c - 2048)];    // [M, G*D]
  else               v = wv[m * 512 + (c - 2560)];    // [M, G*D]
  out[i] = f2bf(v);
}

// ---------------------------------------------------------------------------
// Generic bf16 GEMM: C[Mrows,N] = A[Mrows,K] * B[K,N], f32 accumulate.
// Block tile 128x128, K-step 32, 256 threads = 8 wave32 waves (2x4),
// each wave computes a 64x32 subtile = 4x2 WMMA accumulators.
// A tile staged via GLOBAL_LOAD_ASYNC_TO_LDS_B128; B tile load+transpose
// through VGPRs (scatter); next K-step prefetched with global_prefetch_b8.
// ---------------------------------------------------------------------------
template <bool BF16OUT>
__global__ __launch_bounds__(256) void gemm_bf16_kernel(
    const bf16* __restrict__ A, const bf16* __restrict__ B, void* __restrict__ Cout,
    int Mrows, int N, int K) {
  __shared__ bf16 As[128 * LDP];   // [row][k] row-major, padded
  __shared__ bf16 Bt[128 * LDP];   // [n][k]  transposed, padded

  const int tid  = threadIdx.x;
  const int lane = tid & 31;
  const int wave = tid >> 5;
  const int wm   = wave >> 2;      // 0..1
  const int wn   = wave & 3;       // 0..3
  const int bm   = blockIdx.x * 128;
  const int bn   = blockIdx.y * 128;

  // Per-thread tile-copy coordinates (two rounds of 256 threads each)
  const int ar0 = tid >> 2;                // A rows 0..63 (round 0) / 64..127
  const int ac  = (tid & 3) << 3;          // A col 0,8,16,24
  const int bk0 = tid >> 4;                // B k 0..15 (round 0) / 16..31
  const int bg  = (tid & 15) << 3;         // B n-group start 0..120

  CF acc[4][2] = {};

  for (int k0 = 0; k0 < K; k0 += 32) {
    __syncthreads();
    // --- A tile 128x32: async memory->LDS copy (no VGPR round trip) ---
    async_copy_b128(A + (size_t)(bm + ar0) * K + k0 + ac, &As[ar0 * LDP + ac]);
    async_copy_b128(A + (size_t)(bm + 64 + ar0) * K + k0 + ac,
                    &As[(64 + ar0) * LDP + ac]);
    // --- B tile 32x128: load both rounds, then scatter-transpose to Bt[n][k]
    uint4 d0 = *(const uint4*)(B + (size_t)(k0 + bk0) * N + bn + bg);
    uint4 d1 = *(const uint4*)(B + (size_t)(k0 + 16 + bk0) * N + bn + bg);
    // Prefetch next K-step while stores drain
    if (k0 + 32 < K) {
      __builtin_prefetch(A + (size_t)(bm + ar0) * K + k0 + 32 + ac, 0, 3);
      __builtin_prefetch(B + (size_t)(k0 + 32 + bk0) * N + bn + bg, 0, 3);
    }
    {
      const bf16* e0 = (const bf16*)&d0;
      const bf16* e1 = (const bf16*)&d1;
#pragma unroll
      for (int j = 0; j < 8; ++j) Bt[(bg + j) * LDP + bk0] = e0[j];
#pragma unroll
      for (int j = 0; j < 8; ++j) Bt[(bg + j) * LDP + 16 + bk0] = e1[j];
    }
    wait_async();
    __syncthreads();

    bf16x16 af[4], bfb[2];
#pragma unroll
    for (int fm = 0; fm < 4; ++fm)
      af[fm] = load_frag_a(&As[(wm * 64 + fm * 16) * LDP], LDP, lane);
#pragma unroll
    for (int fn = 0; fn < 2; ++fn)
      bfb[fn] = load_frag_bt(&Bt[(wn * 32 + fn * 16) * LDP], LDP, lane);
#pragma unroll
    for (int fm = 0; fm < 4; ++fm)
#pragma unroll
      for (int fn = 0; fn < 2; ++fn)
        acc[fm][fn].v = wmma_bf16(af[fm], bfb[fn], acc[fm][fn].v);
  }

  // Epilogue: C layout lane holds col n=lane&15, rows r + 8*(lane>>4)
  const int half = lane >> 4, n = lane & 15;
#pragma unroll
  for (int fm = 0; fm < 4; ++fm)
#pragma unroll
    for (int fn = 0; fn < 2; ++fn)
#pragma unroll
      for (int r = 0; r < 8; ++r) {
        size_t row = (size_t)bm + wm * 64 + fm * 16 + half * 8 + r;
        size_t col = (size_t)bn + wn * 32 + fn * 16 + n;
        if (BF16OUT)
          ((bf16*)Cout)[row * N + col] = f2bf(acc[fm][fn].f[r]);
        else
          ((float*)Cout)[row * N + col] = acc[fm][fn].f[r];
      }
}

// ---------------------------------------------------------------------------
// RoPE (half-rotation) on q and k, in place on QKV[BT, NQKV].
// q additionally gets the full muP scale 1/D (= mult^2).
// ---------------------------------------------------------------------------
__global__ void rope_kernel(bf16* __restrict__ QKV) {
  long idx = (long)blockIdx.x * blockDim.x + threadIdx.x;  // BT * 20 * 64
  if (idx >= (long)BT * 20 * 64) return;
  int j   = (int)(idx & 63);               // rotation pair 0..63
  int hs  = (int)((idx >> 6) % 20);        // 0..15 q heads, 16..19 k heads
  int row = (int)(idx / (64 * 20));        // b*T + t
  int t   = row & (KT - 1);
  bool isq = hs < 16;
  int cb = isq ? hs * KD : 2048 + (hs - 16) * KD;
  bf16* p = QKV + (size_t)row * NQKV + cb;
  float ev = bf2f(p[j]);
  float od = bf2f(p[j + 64]);
  float freq = __powf(10000.0f, -(float)j / 64.0f);
  float rad = (float)t * freq;
  float s, c;
  __sincosf(rad, &s, &c);
  float re = ev * c - od * s;
  float ro = ev * s + od * c;
  if (isq) { re *= (1.0f / (float)KD); ro *= (1.0f / (float)KD); }
  p[j]      = f2bf(re);
  p[j + 64] = f2bf(ro);
}

// ---------------------------------------------------------------------------
// Flash attention. grid = (T/128 q-tiles, B*G*H heads), 256 threads = 8 waves.
// Each wave owns 16 q rows: Q kept as 4 A-fragments, O as 8 f32 accumulators.
// 32-key K tiles async-copied to LDS; V transposed through VGPRs; online
// softmax in C-layout with shfl_xor row reductions; P staged via wave-private
// LDS to become the A operand of the P*V WMMA.
// ---------------------------------------------------------------------------
__global__ __launch_bounds__(256) void flash_attn_kernel(
    const bf16* __restrict__ QKV, bf16* __restrict__ O2) {
  __shared__ bf16 Ks[32 * LDKS];        // [key][d]   (already the Bt layout for S)
  __shared__ bf16 Vt[128 * LDP];        // [d][key]   (Bt layout for P*V)
  __shared__ bf16 Pst[8 * 16 * LDP];    // per-wave P staging [16][32]

  const int tid  = threadIdx.x;
  const int lane = tid & 31;
  const int wave = tid >> 5;
  const int m    = lane & 15;
  const int half = lane >> 4;

  const int qb   = blockIdx.x;
  const int head = blockIdx.y;          // 0..31
  const int b    = head >> 4;           // / (G*H)
  const int gh   = head & 15;           // g*H + h
  const int g    = gh >> 2;             // / H
  const int q0   = qb * 128;

  const size_t rowQ0 = (size_t)b * KT + q0;
  const int qcol = gh * KD;
  const int kcol = 2048 + g * KD;
  const int vcol = 2560 + g * KD;

  // Per-thread K/V tile copy coordinates
  const int key0 = tid >> 4;            // 0..15 (round 0) / 16..31
  const int gr   = (tid & 15) << 3;     // d-group start 0..120

  // Q fragments for this wave's 16 rows (1/D scale folded in by RoPE)
  const bf16* Qbase = QKV + (rowQ0 + wave * 16) * NQKV + qcol;
  bf16x16 qf[4];
#pragma unroll
  for (int c = 0; c < 4; ++c) qf[c] = load_frag_a(Qbase + c * 32, NQKV, lane);

  CF oacc[8] = {};
  float mrow[8], lrow[8];
#pragma unroll
  for (int r = 0; r < 8; ++r) { mrow[r] = -__builtin_inff(); lrow[r] = 0.0f; }

  const int ntiles = (q0 + 128) / 32;   // causal: keys <= q0+127
  for (int jt = 0; jt < ntiles; ++jt) {
    const int kbase = jt * 32;
    __syncthreads();
    const size_t rowK0 = (size_t)b * KT + kbase;
    // K tile [32 keys x 128 d]: async memory->LDS copy
    async_copy_b128(QKV + (rowK0 + key0) * NQKV + kcol + gr,
                    &Ks[key0 * LDKS + gr]);
    async_copy_b128(QKV + (rowK0 + 16 + key0) * NQKV + kcol + gr,
                    &Ks[(16 + key0) * LDKS + gr]);
    // V tile transposed to Vt[d][key]: both loads first, then scatter
    uint4 dv0 = *(const uint4*)(QKV + (rowK0 + key0) * NQKV + vcol + gr);
    uint4 dv1 = *(const uint4*)(QKV + (rowK0 + 16 + key0) * NQKV + vcol + gr);
    {
      const bf16* e0 = (const bf16*)&dv0;
      const bf16* e1 = (const bf16*)&dv1;
#pragma unroll
      for (int j = 0; j < 8; ++j) Vt[(gr + j) * LDP + key0] = e0[j];
#pragma unroll
      for (int j = 0; j < 8; ++j) Vt[(gr + j) * LDP + 16 + key0] = e1[j];
    }
    wait_async();
    __syncthreads();

    // S = Q * K^T  (16 x 32 per wave)
    CF s[2] = {};
#pragma unroll
    for (int nf = 0; nf < 2; ++nf)
#pragma unroll
      for (int c = 0; c < 4; ++c) {
        bf16x16 kf = load_frag_bt(&Ks[(nf * 16) * LDKS + c * 32], LDKS, lane);
        s[nf].v = wmma_bf16(qf[c], kf, s[nf].v);
      }

    // Causal mask + online softmax (rows r+8*half, col m per n-frag)
    const int qi_base = q0 + wave * 16 + half * 8;
    float mnew[8];
#pragma unroll
    for (int r = 0; r < 8; ++r) {
      int qi = qi_base + r;
#pragma unroll
      for (int nf = 0; nf < 2; ++nf) {
        int kj = kbase + nf * 16 + m;
        if (kj > qi) s[nf].f[r] -= 1e30f;
      }
      float v0 = fmaxf(s[0].f[r], s[1].f[r]);
      v0 = fmaxf(v0, __shfl_xor(v0, 1, 32));
      v0 = fmaxf(v0, __shfl_xor(v0, 2, 32));
      v0 = fmaxf(v0, __shfl_xor(v0, 4, 32));
      v0 = fmaxf(v0, __shfl_xor(v0, 8, 32));
      mnew[r] = fmaxf(mrow[r], v0);
    }
#pragma unroll
    for (int r = 0; r < 8; ++r) {
      float mn = mnew[r];
      float alpha = __expf(mrow[r] - mn);
      mrow[r] = mn;
      float ps = 0.0f;
#pragma unroll
      for (int nf = 0; nf < 2; ++nf) {
        float p = __expf(s[nf].f[r] - mn);
        s[nf].f[r] = p;
        ps += p;
      }
      ps += __shfl_xor(ps, 1, 32);
      ps += __shfl_xor(ps, 2, 32);
      ps += __shfl_xor(ps, 4, 32);
      ps += __shfl_xor(ps, 8, 32);
      lrow[r] = lrow[r] * alpha + ps;
#pragma unroll
      for (int df = 0; df < 8; ++df) oacc[df].f[r] *= alpha;
    }

    // Stage P (C layout -> A layout) through wave-private LDS
    bf16* P = &Pst[wave * 16 * LDP];
#pragma unroll
    for (int nf = 0; nf < 2; ++nf)
#pragma unroll
      for (int r = 0; r < 8; ++r)
        P[(half * 8 + r) * LDP + nf * 16 + m] = f2bf(s[nf].f[r]);

    bf16x16 pf = load_frag_a(P, LDP, lane);
#pragma unroll
    for (int df = 0; df < 8; ++df) {
      bf16x16 vf = load_frag_bt(&Vt[(df * 16) * LDP], LDP, lane);
      oacc[df].v = wmma_bf16(pf, vf, oacc[df].v);
    }
  }

  // Normalize and write O in [b*T+t, (g*H+h)*D + d] layout (ld = 2048)
  const size_t orow0 = rowQ0 + wave * 16 + half * 8;
#pragma unroll
  for (int df = 0; df < 8; ++df)
#pragma unroll
    for (int r = 0; r < 8; ++r) {
      float val = oacc[df].f[r] / lrow[r];
      O2[(orow0 + r) * (size_t)2048 + gh * KD + df * 16 + m] = f2bf(val);
    }
}

// ---------------------------------------------------------------------------
// Host launch. Workspace layout (bytes, ~76 MB total):
//   xb   [4096x2048] bf16 @ 0
//   wqkv [2048x3072] bf16 @ 16777216
//   qkv  [4096x3072] bf16 @ 29360128
//   waob [2048x2048] bf16 @ 54525952
//   o2   [4096x2048] bf16 @ 62914560   (end: 79691776)
// ---------------------------------------------------------------------------
extern "C" void kernel_launch(void* const* d_in, const int* in_sizes, int n_in,
                              void* d_out, int out_size, void* d_ws, size_t ws_size,
                              hipStream_t stream) {
  (void)in_sizes; (void)n_in; (void)out_size; (void)ws_size;
  const float* x    = (const float*)d_in[0];
  const float* w_aq = (const float*)d_in[1];
  const float* w_ak = (const float*)d_in[2];
  const float* w_av = (const float*)d_in[3];
  const float* w_ao = (const float*)d_in[4];
  float* out = (float*)d_out;

  char* ws = (char*)d_ws;
  bf16* xb   = (bf16*)(ws);
  bf16* wqkv = (bf16*)(ws + 16777216);
  bf16* qkv  = (bf16*)(ws + 29360128);
  bf16* waob = (bf16*)(ws + 54525952);
  bf16* o2   = (bf16*)(ws + 62914560);

  cvt_f32_bf16<<<(BT * KM + 255) / 256, 256, 0, stream>>>(x, xb, BT * KM);
  pack_wqkv<<<(KM * NQKV + 255) / 256, 256, 0, stream>>>(w_aq, w_ak, w_av, wqkv);
  cvt_f32_bf16<<<(KM * KM + 255) / 256, 256, 0, stream>>>(w_ao, waob, KM * KM);

  // QKV = X * Wqkv   [4096,3072]
  gemm_bf16_kernel<true><<<dim3(BT / 128, NQKV / 128), 256, 0, stream>>>(
      xb, wqkv, (void*)qkv, BT, NQKV, KM);

  // RoPE on q (with 1/D scale) and k, in place
  rope_kernel<<<(BT * 20 * 64 + 255) / 256, 256, 0, stream>>>(qkv);

  // Causal flash attention -> O2 [4096,2048] bf16
  flash_attn_kernel<<<dim3(KT / 128, KB * KG * KH), 256, 0, stream>>>(qkv, o2);

  // R = O2 * Wao -> f32 output [4096,2048]
  gemm_bf16_kernel<false><<<dim3(BT / 128, KM / 128), 256, 0, stream>>>(
      o2, waob, (void*)out, BT, KM, KM);
}
